// ExemplarAttention_18846316495263
// MI455X (gfx1250) — matrix operand order repaired and checked by reading, not compiled, not run on hardware
//
#include <hip/hip_runtime.h>
#include <math.h>
#include <stdint.h>

// ---------------------------------------------------------------------------
// ExemplarAttention fused kernel for gfx1250 (MI455X), wave32 + WMMA f16.
//   cross GEMM: v_wmma_f32_16x16x32_f16 (A resident in VGPRs, B from LDS),
//   fused exp(-beta*d) epilogue + per-class LDS binning (ds_add_f32),
//   exemplar tiles staged via global_load_async_to_lds_b128 (ASYNCcnt),
//   double-buffered so the prefetch overlaps the WMMA stream.
// ---------------------------------------------------------------------------

typedef __attribute__((ext_vector_type(16))) _Float16 v16h;
typedef __attribute__((ext_vector_type(8)))  _Float16 v8h;
typedef __attribute__((ext_vector_type(4)))  _Float16 v4h;
typedef __attribute__((ext_vector_type(8)))  float    v8f;

#define EPS_F 1e-9f

constexpr int kB = 1024;     // batch
constexpr int kN = 16384;    // exemplars
constexpr int kD = 512;      // feature dim
constexpr int kC = 10;       // classes

constexpr int ROWS_WG = 64;      // x-rows per workgroup (4 waves x 16)
constexpr int NCHUNK  = 512;     // exemplar columns per workgroup
constexpr int NTILES  = NCHUNK / 16;
constexpr int KSTEPS  = kD / 32;
constexpr int LDSTR   = kD + 8;  // padded f16 row stride (1040 B = 65 x 16 B)

union HFrag { v16h v; v8h h[2]; };

__device__ __forceinline__ float softplus_f(float z) {
    return (z > 20.0f) ? z : log1pf(expf(z));
}

__device__ __forceinline__ void wait_asynccnt0() {
#if __has_builtin(__builtin_amdgcn_s_wait_asynccnt)
    __builtin_amdgcn_s_wait_asynccnt(0);
#else
    asm volatile("s_wait_asynccnt 0x0" ::: "memory");
#endif
}

// 16-byte async global->LDS copy with shared immediate offset (IOFFSET is
// added to both the LDS and global addresses, ISA 10.7/15.18).
// ldsOff = low 32 bits of generic shared pointer == LDS byte address.
template <int OFF>
__device__ __forceinline__ void async_cp16(uint32_t ldsOff, const void* g) {
    asm volatile("global_load_async_to_lds_b128 %0, %1, off offset:%2"
                 :: "v"(ldsOff), "v"((uint64_t)(uintptr_t)g), "i"(OFF)
                 : "memory");
}

// ---------------------------------------------------------------------------
// Kernel 1: constrained params. softmax(w_u)+EPS -> w[D]; softplus scalars.
// ---------------------------------------------------------------------------
__global__ __launch_bounds__(512)
void prep_params(const float* __restrict__ wu,
                 const float* __restrict__ gu,
                 const float* __restrict__ bu,
                 float* __restrict__ w_out,
                 float* __restrict__ scal /* [0]=beta [1]=gamma */) {
    __shared__ float red[512];
    const int t = threadIdx.x;
    const float v = wu[t];

    red[t] = v;
    __syncthreads();
    for (int s = 256; s > 0; s >>= 1) {
        if (t < s) red[t] = fmaxf(red[t], red[t + s]);
        __syncthreads();
    }
    const float mx = red[0];
    __syncthreads();

    const float e = expf(v - mx);
    red[t] = e;
    __syncthreads();
    for (int s = 256; s > 0; s >>= 1) {
        if (t < s) red[t] += red[t + s];
        __syncthreads();
    }
    w_out[t] = e / red[0] + EPS_F;

    if (t == 0) {
        scal[0] = softplus_f(bu[0]) + EPS_F;   // beta
        scal[1] = softplus_f(gu[0]) + EPS_F;   // gamma
    }
}

// ---------------------------------------------------------------------------
// Kernel 2: xw_f16[b,k] = f16(x[b,k]*w[k]); x2w[b] = sum_k w_k * x^2.
// ---------------------------------------------------------------------------
__global__ __launch_bounds__(512)
void prep_xw(const float* __restrict__ x,
             const float* __restrict__ w,
             _Float16* __restrict__ xwh,
             float* __restrict__ x2w) {
    __shared__ float red[512];
    const int b = blockIdx.x;
    const int k = threadIdx.x;
    const float wv = w[k];
    const float xv = x[(size_t)b * kD + k];
    xwh[(size_t)b * kD + k] = (_Float16)(xv * wv);
    red[k] = xv * xv * wv;
    __syncthreads();
    for (int s = 256; s > 0; s >>= 1) {
        if (k < s) red[k] += red[k + s];
        __syncthreads();
    }
    if (k == 0) x2w[b] = red[0];
}

// ---------------------------------------------------------------------------
// Kernel 3: e2w[n] = sum_k w_k * e^2 ; optionally eh[n,k] = f16(e[n,k]).
// ---------------------------------------------------------------------------
__global__ __launch_bounds__(512)
void prep_e2w(const float* __restrict__ ex,
              const float* __restrict__ w,
              float* __restrict__ e2w,
              _Float16* __restrict__ eh /* may be null */) {
    __shared__ float red[512];
    const int n = blockIdx.x;
    const int k = threadIdx.x;
    const float ev = ex[(size_t)n * kD + k];
    if (eh) eh[(size_t)n * kD + k] = (_Float16)ev;
    red[k] = ev * ev * w[k];
    __syncthreads();
    for (int s = 256; s > 0; s >>= 1) {
        if (k < s) red[k] += red[k + s];
        __syncthreads();
    }
    if (k == 0) e2w[n] = red[0];
}

__global__ void zero_f32(float* __restrict__ p, int n) {
    int i = blockIdx.x * blockDim.x + threadIdx.x;
    if (i < n) p[i] = 0.0f;
}

// ---------------------------------------------------------------------------
// Shared epilogue: d = x2w + e2w - 2*cross ; sim = exp(-beta d); bin by label.
// ---------------------------------------------------------------------------
__device__ __forceinline__
void tile_epilogue(const v8f& acc, const float* x2wr, float ew, int lab,
                   float negBetaLog2e, int wave, int laneHalf,
                   float* cls) {
#pragma unroll
    for (int r = 0; r < 8; ++r) {
        const float d   = x2wr[r] + ew - 2.0f * acc[r];
        const float sim = exp2f(negBetaLog2e * d);
        const int   lr  = wave * 16 + r + 8 * laneHalf;
        atomicAdd(&cls[lr * kC + lab], sim);              // ds_add_f32
    }
}

// Load all 16 A fragments (this wave's 16x512 f16 slice) into registers.
// A 16x32 layout: lanes<16 hold K 0-7 & 16-23; lanes>=16 hold K 8-15 & 24-31.
__device__ __forceinline__
void load_a_frags(HFrag* aF, const _Float16* aBase, int laneHalf) {
#pragma unroll
    for (int ks = 0; ks < KSTEPS; ++ks) {
        aF[ks].h[0] = *(const v8h*)(aBase + ks * 32 + 8 * laneHalf);
        aF[ks].h[1] = *(const v8h*)(aBase + ks * 32 + 16 + 8 * laneHalf);
    }
}

// B fragment from LDS. B 32x16: lane = column, lanes<16 K 0-7 & 16-23, >=16 K 8-15 & 24-31.
__device__ __forceinline__
v16h load_b_frag(const _Float16* bBase, int k0, int laneHalf) {
    HFrag b;
    const _Float16* bp = bBase + k0 + 8 * laneHalf;
    b.h[0] = *(const v8h*)(bp);
    b.h[1] = *(const v8h*)(bp + 16);
    return b.v;
}

// ---------------------------------------------------------------------------
// Kernel 4a: fused WMMA GEMM, async-staged (needs eh f16 buffer).
// grid = (N/NCHUNK, B/ROWS_WG), block = 128 (4 waves).
// ---------------------------------------------------------------------------
__global__ __launch_bounds__(128)
void gemm_fused_async(const _Float16* __restrict__ xwh,
                      const _Float16* __restrict__ eh,
                      const int* __restrict__ labels,
                      const float* __restrict__ x2w,
                      const float* __restrict__ e2w,
                      const float* __restrict__ scal,
                      float* __restrict__ class_sum) {
    __shared__ _Float16 eT[2][16 * LDSTR];
    __shared__ float    cls[ROWS_WG * kC];

    const int t        = threadIdx.x;
    const int wave     = t >> 5;
    const int lane     = t & 31;
    const int laneHalf = lane >> 4;
    const int l15      = lane & 15;

    const int row0    = blockIdx.y * ROWS_WG;
    const int waveRow = row0 + wave * 16;
    const int nBase   = blockIdx.x * NCHUNK;

    for (int i = t; i < ROWS_WG * kC; i += 128) cls[i] = 0.0f;

    const float beta = scal[0];
    const float negBetaLog2e = -beta * 1.44269504088896f;

    float x2wr[8];
#pragma unroll
    for (int r = 0; r < 8; ++r)
        x2wr[r] = x2w[waveRow + r + 8 * laneHalf];

    // A fragments resident in VGPRs for the whole kernel (register values are
    // unaffected by the asm "memory" clobber of the async staging below).
    HFrag aF[KSTEPS];
    load_a_frags(aF, xwh + (size_t)(waveRow + l15) * kD, laneHalf);

    // Staging: wave w copies tile rows [4w, 4w+4); 8 b128/lane per tile.
    // Shared linear part i*512 goes in the instruction IOFFSET (applied to
    // both sides); the +16 B/row LDS pad lives in precomputed row bases.
    const char* gWave0 =
        (const char*)(eh) + (size_t)(4 * wave) * kD * 2 + lane * 16;
    uint32_t lRow[2][4];
#pragma unroll
    for (int bfi = 0; bfi < 2; ++bfi) {
        const uint32_t base = (uint32_t)(uintptr_t)(
            (char*)&eT[bfi][0] + 4 * wave * (LDSTR * 2) + lane * 16);
#pragma unroll
        for (int r = 0; r < 4; ++r) lRow[bfi][r] = base + r * 16;
    }

    auto stage = [&](int n0, int bfi) {
        const char* g = gWave0 + (size_t)n0 * kD * 2;
#pragma unroll
        for (int i = 0; i < 8; ++i) {
            constexpr int kStep = 512;
            switch (i) {   // constant IOFFSET per copy
            case 0: async_cp16<0 * kStep>(lRow[0][0] + (bfi ? sizeof(eT[0]) : 0), g); break;
            case 1: async_cp16<1 * kStep>(lRow[0][0] + (bfi ? sizeof(eT[0]) : 0), g); break;
            case 2: async_cp16<2 * kStep>(lRow[0][1] + (bfi ? sizeof(eT[0]) : 0), g); break;
            case 3: async_cp16<3 * kStep>(lRow[0][1] + (bfi ? sizeof(eT[0]) : 0), g); break;
            case 4: async_cp16<4 * kStep>(lRow[0][2] + (bfi ? sizeof(eT[0]) : 0), g); break;
            case 5: async_cp16<5 * kStep>(lRow[0][2] + (bfi ? sizeof(eT[0]) : 0), g); break;
            case 6: async_cp16<6 * kStep>(lRow[0][3] + (bfi ? sizeof(eT[0]) : 0), g); break;
            case 7: async_cp16<7 * kStep>(lRow[0][3] + (bfi ? sizeof(eT[0]) : 0), g); break;
            }
        }
    };

    // Prologue: fill buffer 0 with tile 0.
    stage(nBase, 0);
    wait_asynccnt0();
    __syncthreads();

    for (int ct = 0; ct < NTILES; ++ct) {
        const int cur = ct & 1;
        if (ct + 1 < NTILES) stage(nBase + (ct + 1) * 16, cur ^ 1);

        const _Float16* bBase = &eT[cur][l15 * LDSTR];

        v8f acc = {};
#pragma unroll
        for (int ks = 0; ks < KSTEPS; ++ks) {
            const v16h bV = load_b_frag(bBase, ks * 32, laneHalf);
            acc = __builtin_amdgcn_wmma_f32_16x16x32_f16(
                false, aF[ks].v, false, bV, (short)0, acc, false, false);
        }

        const int n = nBase + ct * 16 + l15;
        tile_epilogue(acc, x2wr, e2w[n], labels[n], negBetaLog2e,
                      wave, laneHalf, cls);

        wait_asynccnt0();   // our prefetch stores landed in LDS
        __syncthreads();    // everyone's stores landed; tile `cur` consumed
    }

    for (int i = t; i < ROWS_WG * kC; i += 128) {
        atomicAdd(&class_sum[(size_t)(row0 + i / kC) * kC + (i % kC)], cls[i]);
    }
}

// ---------------------------------------------------------------------------
// Kernel 4b: fallback (small d_ws): stage via f32 load + cvt + ds_store.
// ---------------------------------------------------------------------------
__global__ __launch_bounds__(128)
void gemm_fused_cvt(const _Float16* __restrict__ xwh,
                    const float* __restrict__ ex,
                    const int* __restrict__ labels,
                    const float* __restrict__ x2w,
                    const float* __restrict__ e2w,
                    const float* __restrict__ scal,
                    float* __restrict__ class_sum) {
    __shared__ _Float16 eT[16 * LDSTR];
    __shared__ float    cls[ROWS_WG * kC];

    const int t        = threadIdx.x;
    const int wave     = t >> 5;
    const int lane     = t & 31;
    const int laneHalf = lane >> 4;
    const int l15      = lane & 15;

    const int row0    = blockIdx.y * ROWS_WG;
    const int waveRow = row0 + wave * 16;
    const int nBase   = blockIdx.x * NCHUNK;

    for (int i = t; i < ROWS_WG * kC; i += 128) cls[i] = 0.0f;

    const float beta = scal[0];
    const float negBetaLog2e = -beta * 1.44269504088896f;

    float x2wr[8];
#pragma unroll
    for (int r = 0; r < 8; ++r)
        x2wr[r] = x2w[waveRow + r + 8 * laneHalf];

    HFrag aF[KSTEPS];
    load_a_frags(aF, xwh + (size_t)(waveRow + l15) * kD, laneHalf);

    const _Float16* bBase = &eT[l15 * LDSTR];

    for (int ct = 0; ct < NTILES; ++ct) {
        const int n0 = nBase + ct * 16;

        __syncthreads();
#pragma unroll
        for (int j = 0; j < 16; ++j) {
            const float4 ev = *(const float4*)(ex + (size_t)(n0 + j) * kD + t * 4);
            v4h hv;
            hv[0] = (_Float16)ev.x;
            hv[1] = (_Float16)ev.y;
            hv[2] = (_Float16)ev.z;
            hv[3] = (_Float16)ev.w;
            *(v4h*)(&eT[j * LDSTR + t * 4]) = hv;
        }
        __syncthreads();

        v8f acc = {};
#pragma unroll
        for (int ks = 0; ks < KSTEPS; ++ks) {
            const v16h bV = load_b_frag(bBase, ks * 32, laneHalf);
            acc = __builtin_amdgcn_wmma_f32_16x16x32_f16(
                false, aF[ks].v, false, bV, (short)0, acc, false, false);
        }

        const int n = n0 + l15;
        tile_epilogue(acc, x2wr, e2w[n], labels[n], negBetaLog2e,
                      wave, laneHalf, cls);
    }

    __syncthreads();
    for (int i = t; i < ROWS_WG * kC; i += 128) {
        atomicAdd(&class_sum[(size_t)(row0 + i / kC) * kC + (i % kC)], cls[i]);
    }
}

// ---------------------------------------------------------------------------
// Kernel 5: logits = gamma * log(class_sum + EPS)
// ---------------------------------------------------------------------------
__global__ void finalize(const float* __restrict__ csum,
                         const float* __restrict__ scal,
                         float* __restrict__ out, int n) {
    int i = blockIdx.x * blockDim.x + threadIdx.x;
    if (i < n) out[i] = scal[1] * logf(csum[i] + EPS_F);
}

// ---------------------------------------------------------------------------
// Host launcher. Inputs (setup_inputs order):
//   0: x (B,D) f32   1: ex_feats (N,D) f32   2: ex_labels (N,) i32
//   3: w_unconstrained (D,) f32   4: gamma_u f32[1]   5: beta_u f32[1]
// Output: logits (B,10) f32.
// ---------------------------------------------------------------------------
extern "C" void kernel_launch(void* const* d_in, const int* in_sizes, int n_in,
                              void* d_out, int out_size, void* d_ws, size_t ws_size,
                              hipStream_t stream) {
    const float* x      = (const float*)d_in[0];
    const float* ex     = (const float*)d_in[1];
    const int*   labels = (const int*)d_in[2];
    const float* wu     = (const float*)d_in[3];
    const float* gu     = (const float*)d_in[4];
    const float* bu     = (const float*)d_in[5];
    float* out = (float*)d_out;

    char* ws = (char*)d_ws;
    float*    w    = (float*)(ws + 0);          // 512 f32
    float*    scal = (float*)(ws + 2048);       // beta, gamma
    float*    x2w  = (float*)(ws + 2560);       // 1024 f32
    float*    e2w  = (float*)(ws + 6656);       // 16384 f32
    float*    csum = (float*)(ws + 72192);      // 10240 f32
    _Float16* xwh  = (_Float16*)(ws + 113152);  // B*D f16 (1 MB)
    _Float16* eh   = (_Float16*)(ws + 1161728); // N*D f16 (16 MB)

    const size_t needAsync = 1161728ULL + (size_t)kN * kD * 2;  // ~17.2 MB
    const bool useAsync = (ws_size >= needAsync);

    prep_params<<<1, 512, 0, stream>>>(wu, gu, bu, w, scal);
    prep_xw   <<<kB, 512, 0, stream>>>(x, w, xwh, x2w);
    prep_e2w  <<<kN, 512, 0, stream>>>(ex, w, e2w, useAsync ? eh : nullptr);
    zero_f32  <<<(kB * kC + 255) / 256, 256, 0, stream>>>(csum, kB * kC);

    dim3 grid(kN / NCHUNK, kB / ROWS_WG);       // (32, 16) workgroups
    if (useAsync) {
        gemm_fused_async<<<grid, 128, 0, stream>>>(xwh, eh, labels, x2w, e2w,
                                                   scal, csum);
    } else {
        gemm_fused_cvt<<<grid, 128, 0, stream>>>(xwh, ex, labels, x2w, e2w,
                                                 scal, csum);
    }

    finalize<<<(kB * kC + 255) / 256, 256, 0, stream>>>(csum, scal, out, kB * kC);
}
